// Model_HP_59571196395834
// MI455X (gfx1250) — matrix operand order repaired and checked by reading, not compiled, not run on hardware
//
#include <hip/hip_runtime.h>

#define NNODES 50000
#define NEDGES 800000
#define MROWS  12500
#define INDIM  256
#define HID    128
#define OUTD   64
#define EPSV   1e-5f
#define SLOPE  0.1f

typedef __attribute__((ext_vector_type(16))) _Float16 v16h;
typedef __attribute__((ext_vector_type(8)))  float    v8f;

// ---------------------------------------------------------------------------
// A fragment 16x32 f16 from row-major f32 [*, lda]  (ISA 7.12.2 layout).
// Per-lane K footprint is two contiguous 8-float runs -> 4x float4 loads.
// AFFINE: fused per-channel scale/shift (BatchNorm). ROWSCALE: per-row factor
// (deg_inv for mean aggregation).
// ---------------------------------------------------------------------------
template <bool AFFINE, bool ROWSCALE>
__device__ __forceinline__ v16h load_frag_a(const float* A, int lda, int r0, int k0,
                                            const float* scale, const float* shift,
                                            const float* rowinv) {
  const int lane = threadIdx.x & 31;
  const int row  = r0 + (lane & 15);
  const int kg   = (lane >> 4) << 3;                 // 0 or 8
  const float* rp = A + (size_t)row * lda + k0 + kg;

  float vals[16];
  *(float4*)(vals + 0)  = *(const float4*)(rp + 0);
  *(float4*)(vals + 4)  = *(const float4*)(rp + 4);
  *(float4*)(vals + 8)  = *(const float4*)(rp + 16);
  *(float4*)(vals + 12) = *(const float4*)(rp + 20);

  if (AFFINE) {
    const float* sp = scale + k0 + kg;
    const float* hp = shift + k0 + kg;
    float sc[16], sh[16];
    *(float4*)(sc + 0)  = *(const float4*)(sp + 0);
    *(float4*)(sc + 4)  = *(const float4*)(sp + 4);
    *(float4*)(sc + 8)  = *(const float4*)(sp + 16);
    *(float4*)(sc + 12) = *(const float4*)(sp + 20);
    *(float4*)(sh + 0)  = *(const float4*)(hp + 0);
    *(float4*)(sh + 4)  = *(const float4*)(hp + 4);
    *(float4*)(sh + 8)  = *(const float4*)(hp + 16);
    *(float4*)(sh + 12) = *(const float4*)(hp + 20);
#pragma unroll
    for (int e = 0; e < 16; ++e) vals[e] = vals[e] * sc[e] + sh[e];
  }

  float rs = 1.0f;
  if (ROWSCALE) rs = rowinv[row];

  v16h out;
#pragma unroll
  for (int e = 0; e < 16; ++e) out[e] = (_Float16)(vals[e] * rs);
  return out;
}

// ---------------------------------------------------------------------------
// Pre-pack a row-major f32 weight [kdim, ncols] into per-lane WMMA B
// fragments (f16): frag index [(kt*nTiles + nt)*32 + lane][0..15].
// gridDim = (kdim/32, ncols/16), block = 32.
// ---------------------------------------------------------------------------
__global__ void pack_b_kernel(const float* W, _Float16* outp, int ncols) {
  const int kt   = blockIdx.x;
  const int nt   = blockIdx.y;
  const int lane = threadIdx.x;
  const int col  = nt * 16 + (lane & 15);
  const int kh   = (lane >> 4) << 3;                 // 0 or 8
  v16h frag;
#pragma unroll
  for (int e = 0; e < 16; ++e)
    frag[e] = (_Float16)W[(size_t)(kt * 32 + kh + (e < 8 ? e : e + 8)) * ncols + col];
  *(v16h*)(outp + ((size_t)(kt * gridDim.y + nt) * 32 + lane) * 16) = frag;
}

// ---------------------------------------------------------------------------
// out = [leaky]( affine(A1) @ B1  +  (rowinv*A2) @ B2  + bias )
// grid.x = row tiles (16 rows), blockDim = 32 * (ldo/16): one wave per col
// tile. B operands pre-packed; their load is one aligned 32B vector per lane.
// ---------------------------------------------------------------------------
template <int KDIM, bool AFFINE, bool HAS_A2, bool LEAKY>
__global__ void gemm_wmma_kernel(const float* A1, const float* scale1, const float* shift1,
                                 const float* A2, const float* rowinv,
                                 const _Float16* packB1, const _Float16* packB2,
                                 const float* bias, float* outp, int lda, int ldo) {
  const int r0     = blockIdx.x << 4;
  const int wave   = threadIdx.x >> 5;
  const int nTiles = blockDim.x >> 5;
  const int lane   = threadIdx.x & 31;
  const int col    = (wave << 4) + (lane & 15);
  const float bv   = bias[col];
  v8f c = {bv, bv, bv, bv, bv, bv, bv, bv};
#pragma unroll
  for (int k0 = 0; k0 < KDIM; k0 += 32) {
    const int kt = k0 >> 5;
    const v16h a = load_frag_a<AFFINE, false>(A1, lda, r0, k0, scale1, shift1, nullptr);
    const v16h b = *(const v16h*)(packB1 + ((size_t)(kt * nTiles + wave) * 32 + lane) * 16);
    c = __builtin_amdgcn_wmma_f32_16x16x32_f16(false, a, false, b, (short)0, c, false, false);
    if (HAS_A2) {
      const v16h a2 = load_frag_a<false, true>(A2, lda, r0, k0, nullptr, nullptr, rowinv);
      const v16h b2 = *(const v16h*)(packB2 + ((size_t)(kt * nTiles + wave) * 32 + lane) * 16);
      c = __builtin_amdgcn_wmma_f32_16x16x32_f16(false, a2, false, b2, (short)0, c, false, false);
    }
  }
  const int rowbase = r0 + ((lane >> 4) << 3);
#pragma unroll
  for (int r = 0; r < 8; ++r) {
    float v = c[r];
    if (LEAKY) v = (v >= 0.0f) ? v : SLOPE * v;
    outp[(size_t)(rowbase + r) * ldo + col] = v;
  }
}

// ---- BatchNorm statistics (training-mode batch mean/var over N rows) ----
__global__ void bn_stats_kernel(const float* h, float* sum, float* sumsq, int n) {
  const int c    = threadIdx.x & (HID - 1);
  const int half = threadIdx.x >> 7;
  float s = 0.0f, sq = 0.0f;
  for (int r = blockIdx.x * 2 + half; r < n; r += gridDim.x * 2) {
    float v = h[(size_t)r * HID + c];
    s += v; sq += v * v;
  }
  __shared__ float ls[256], lq[256];
  ls[threadIdx.x] = s; lq[threadIdx.x] = sq;
  __syncthreads();
  if (half == 0) {
    atomicAdd(&sum[c],   ls[c] + ls[c + HID]);
    atomicAdd(&sumsq[c], lq[c] + lq[c + HID]);
  }
}

__global__ void bn_finalize_kernel(const float* sum, const float* sumsq,
                                   const float* g, const float* b,
                                   float* scale, float* shift) {
  const int c = threadIdx.x;
  const float mu  = sum[c] * (1.0f / NNODES);
  const float var = sumsq[c] * (1.0f / NNODES) - mu * mu;
  const float rs  = rsqrtf(var + EPSV);
  const float sc  = g[c] * rs;
  scale[c] = sc;
  shift[c] = b[c] - mu * sc;
}

// ---- Histogram / degree / segment counting ----
__global__ void count_kernel(const int* idx, float* cnt, int n) {
  const int i = blockIdx.x * blockDim.x + threadIdx.x;
  if (i < n) atomicAdd(&cnt[idx[i]], 1.0f);
}

__global__ void invclamp_kernel(float* a, int n) {
  const int i = blockIdx.x * blockDim.x + threadIdx.x;
  if (i < n) a[i] = 1.0f / fmaxf(a[i], 1.0f);
}

// ---- Edge gather + scatter-add: agg[dst] += affine(h[src]) (L2-resident) ----
template <bool AFFINE>
__global__ void edge_scatter_kernel(const float* h, const int* src, const int* dst,
                                    const float* scale, const float* shift,
                                    float* agg, int ne) {
  const int e = blockIdx.x * 8 + (threadIdx.x >> 5);
  if (e >= ne) return;
  const int lane = threadIdx.x & 31;
  const int s = src[e], d = dst[e];
  const float4 v = *(const float4*)(h + (size_t)s * HID + lane * 4);
  float vals[4] = {v.x, v.y, v.z, v.w};
  if (AFFINE) {
    const float4 sc = *(const float4*)(scale + lane * 4);
    const float4 sh = *(const float4*)(shift + lane * 4);
    vals[0] = vals[0] * sc.x + sh.x;
    vals[1] = vals[1] * sc.y + sh.y;
    vals[2] = vals[2] * sc.z + sh.z;
    vals[3] = vals[3] * sc.w + sh.w;
  }
#pragma unroll
  for (int j = 0; j < 4; ++j)
    atomicAdd(&agg[(size_t)d * HID + lane * 4 + j], vals[j]);
}

// ---- Row-normalized segment pooling ----
__global__ void pool_scatter_kernel(const float* xp, const int* seg, const float* cinv,
                                    float* y, int n) {
  const int node = blockIdx.x * 4 + (threadIdx.x >> 6);
  if (node >= n) return;
  const int f = threadIdx.x & 63;
  const int s = seg[node];
  const float w = cinv[s];
  atomicAdd(&y[(size_t)s * OUTD + f], xp[(size_t)node * OUTD + f] * w);
}

// ---- MSE reduction ----
__global__ void loss_reduce_kernel(const float* y, const float* x, float* loss, int n) {
  float s = 0.0f;
  for (int i = blockIdx.x * blockDim.x + threadIdx.x; i < n; i += gridDim.x * blockDim.x) {
    const float d = y[i] - x[i];
    s += d * d;
  }
  __shared__ float red[256];
  red[threadIdx.x] = s;
  __syncthreads();
  for (int off = 128; off > 0; off >>= 1) {
    if (threadIdx.x < off) red[threadIdx.x] += red[threadIdx.x + off];
    __syncthreads();
  }
  if (threadIdx.x == 0) atomicAdd(loss, red[0]);
}

__global__ void finalize_kernel(const float* loss, float* outp) {
  const int i = threadIdx.x;
  if (i < 2) outp[i] = loss[i] * (1.0f / (float)(MROWS * OUTD));
}

extern "C" void kernel_launch(void* const* d_in, const int* in_sizes, int n_in,
                              void* d_out, int out_size, void* d_ws, size_t ws_size,
                              hipStream_t stream) {
  (void)in_sizes; (void)n_in; (void)out_size; (void)ws_size;

  char* w = (char*)d_ws;
  auto carve = [&](size_t bytes) -> void* {
    char* p = w;
    w += (bytes + 255) & ~(size_t)255;
    return p;
  };
  float* hbuf0   = (float*)carve((size_t)NNODES * HID * 4);
  float* hbuf1   = (float*)carve((size_t)NNODES * HID * 4);
  float* agg     = (float*)carve((size_t)NNODES * HID * 4);
  float* xp      = (float*)carve((size_t)NNODES * OUTD * 4);
  float* deg     = (float*)carve((size_t)NNODES * 4);
  float* cnt     = (float*)carve((size_t)MROWS * 4);
  float* ybuf    = (float*)carve((size_t)MROWS * OUTD * 4);
  float* bnsum   = (float*)carve(HID * 4);
  float* bnsq    = (float*)carve(HID * 4);
  float* bnscale = (float*)carve(HID * 4);
  float* bnshift = (float*)carve(HID * 4);
  float* loss    = (float*)carve(2 * 4);
  _Float16* pB1  = (_Float16*)carve((size_t)INDIM * HID * 2);  // largest packed weight
  _Float16* pB2  = (_Float16*)carve((size_t)HID * HID * 2);

  hipMemsetAsync(loss, 0, 2 * 4, stream);

  const int rowTiles = NNODES / 16;  // 3125 (N divisible by 16)

  for (int p = 0; p < 2; ++p) {
    const float* feat = (const float*)d_in[p];
    const float* xtgt = (const float*)d_in[2 + p];
    const int*   src  = (const int*)d_in[4 + 2 * p];
    const int*   dst  = (const int*)d_in[5 + 2 * p];
    const int*   seg  = (const int*)d_in[8 + p];
    const int pb = 10 + 12 * p;
    const float* W_mlp = (const float*)d_in[pb + 0];
    const float* b_mlp = (const float*)d_in[pb + 1];
    const float* bn_g  = (const float*)d_in[pb + 2];
    const float* bn_b  = (const float*)d_in[pb + 3];
    const float* Wl0   = (const float*)d_in[pb + 4];
    const float* Wr0   = (const float*)d_in[pb + 5];
    const float* b0    = (const float*)d_in[pb + 6];
    const float* Wl1   = (const float*)d_in[pb + 7];
    const float* Wr1   = (const float*)d_in[pb + 8];
    const float* b1    = (const float*)d_in[pb + 9];
    const float* Wp    = (const float*)d_in[pb + 10];
    const float* bp    = (const float*)d_in[pb + 11];

    // encode: hbuf0 = leaky(feat @ W_mlp + b_mlp)
    pack_b_kernel<<<dim3(INDIM / 32, HID / 16), 32, 0, stream>>>(W_mlp, pB1, HID);
    gemm_wmma_kernel<INDIM, false, false, true><<<rowTiles, 32 * (HID / 16), 0, stream>>>(
        feat, nullptr, nullptr, nullptr, nullptr, pB1, nullptr, b_mlp, hbuf0, INDIM, HID);

    // BatchNorm batch statistics -> fused affine (scale/shift)
    hipMemsetAsync(bnsum, 0, HID * 4, stream);
    hipMemsetAsync(bnsq, 0, HID * 4, stream);
    bn_stats_kernel<<<256, 256, 0, stream>>>(hbuf0, bnsum, bnsq, NNODES);
    bn_finalize_kernel<<<1, HID, 0, stream>>>(bnsum, bnsq, bn_g, bn_b, bnscale, bnshift);

    // in-degrees -> deg_inv (in place)
    hipMemsetAsync(deg, 0, NNODES * 4, stream);
    count_kernel<<<(NEDGES + 255) / 256, 256, 0, stream>>>(dst, deg, NEDGES);
    invclamp_kernel<<<(NNODES + 255) / 256, 256, 0, stream>>>(deg, NNODES);

    // SAGE layer 0 (BN affine fused into scatter and GEMM A-load)
    hipMemsetAsync(agg, 0, (size_t)NNODES * HID * 4, stream);
    edge_scatter_kernel<true><<<(NEDGES + 7) / 8, 256, 0, stream>>>(
        hbuf0, src, dst, bnscale, bnshift, agg, NEDGES);
    pack_b_kernel<<<dim3(HID / 32, HID / 16), 32, 0, stream>>>(Wl0, pB1, HID);
    pack_b_kernel<<<dim3(HID / 32, HID / 16), 32, 0, stream>>>(Wr0, pB2, HID);
    gemm_wmma_kernel<HID, true, true, true><<<rowTiles, 32 * (HID / 16), 0, stream>>>(
        hbuf0, bnscale, bnshift, agg, deg, pB1, pB2, b0, hbuf1, HID, HID);

    // SAGE layer 1
    hipMemsetAsync(agg, 0, (size_t)NNODES * HID * 4, stream);
    edge_scatter_kernel<false><<<(NEDGES + 7) / 8, 256, 0, stream>>>(
        hbuf1, src, dst, nullptr, nullptr, agg, NEDGES);
    pack_b_kernel<<<dim3(HID / 32, HID / 16), 32, 0, stream>>>(Wl1, pB1, HID);
    pack_b_kernel<<<dim3(HID / 32, HID / 16), 32, 0, stream>>>(Wr1, pB2, HID);
    gemm_wmma_kernel<HID, false, true, true><<<rowTiles, 32 * (HID / 16), 0, stream>>>(
        hbuf1, nullptr, nullptr, agg, deg, pB1, pB2, b1, hbuf0, HID, HID);

    // projection: xp = hbuf0 @ Wp + bp
    pack_b_kernel<<<dim3(HID / 32, OUTD / 16), 32, 0, stream>>>(Wp, pB1, OUTD);
    gemm_wmma_kernel<HID, false, false, false><<<rowTiles, 32 * (OUTD / 16), 0, stream>>>(
        hbuf0, nullptr, nullptr, nullptr, nullptr, pB1, nullptr, bp, xp, HID, OUTD);

    // row-normalized pooling to M rows
    hipMemsetAsync(cnt, 0, MROWS * 4, stream);
    count_kernel<<<(NNODES + 255) / 256, 256, 0, stream>>>(seg, cnt, NNODES);
    invclamp_kernel<<<(MROWS + 255) / 256, 256, 0, stream>>>(cnt, MROWS);
    hipMemsetAsync(ybuf, 0, (size_t)MROWS * OUTD * 4, stream);
    pool_scatter_kernel<<<(NNODES + 3) / 4, 256, 0, stream>>>(xp, seg, cnt, ybuf, NNODES);

    // MSE
    loss_reduce_kernel<<<512, 256, 0, stream>>>(ybuf, xtgt, loss + p, MROWS * OUTD);
  }

  finalize_kernel<<<1, 32, 0, stream>>>(loss, (float*)d_out);
}
